// ExactSphericalYatAttention_43026982371799
// MI455X (gfx1250) — matrix operand order repaired and checked by reading, not compiled, not run on hardware
//
#include <hip/hip_runtime.h>

// ===== problem constants =====
#define EMBED  1024
#define HEADS  16
#define HDIM   64
#define SEQ    2048
#define BATCH  2
#define MROWS  (BATCH * SEQ)   // 4096
#define NQKV   (3 * EMBED)     // 3072
#define CCONST (2.0f + 1e-6f)

typedef _Float16 h16;
typedef __bf16   bf16;
typedef __attribute__((ext_vector_type(16))) _Float16 v16h;
typedef __attribute__((ext_vector_type(16))) __bf16   v16bf;
typedef __attribute__((ext_vector_type(8)))  float    v8f;
typedef __attribute__((ext_vector_type(4)))  float    f32x4;

union FragH  { v16h  v; f32x4 f4[2]; };
union FragBF { v16bf v; f32x4 f4[2]; };

// ---- Tensor Data Mover availability (probe-documented builtins) ----
#if defined(__has_builtin)
#if __has_builtin(__builtin_amdgcn_tensor_load_to_lds) && \
    __has_builtin(__builtin_amdgcn_s_wait_tensorcnt)
#define USE_TDM 1
#endif
#endif
#ifndef USE_TDM
#define USE_TDM 0
#endif

#if USE_TDM
typedef unsigned int u32x4 __attribute__((ext_vector_type(4)));
typedef int          i32x4 __attribute__((ext_vector_type(4)));
typedef int          i32x8 __attribute__((ext_vector_type(8)));

// Issue one TDM 2D tile load: 64x64 elements of 2B, rows padded in LDS by
// 4 DWORDs every 32 DWORDs (-> 144B row stride == our 72-half LDS layout).
// Descriptor packing per CDNA5 ISA.
__device__ __forceinline__ void tdm_load_tile_2d(
    const void* gaddr, unsigned ldsOff,
    unsigned tensor_d0, unsigned tensor_d1, unsigned d0_stride) {
    unsigned long long ga = (unsigned long long)(size_t)gaddr;
    u32x4 g0;
    g0[0] = 1u;                                            // count=1, user mode
    g0[1] = ldsOff;                                        // lds_addr (bytes)
    g0[2] = (unsigned)(ga & 0xffffffffu);                  // global_addr[31:0]
    g0[3] = (unsigned)((ga >> 32) & 0x01ffffffu) | (2u << 30); // addr[56:32]|type=2
    i32x8 g1;
    g1[0] = (1 << 16)    // data_size = 2 bytes
          | (1 << 20)    // pad_enable
          | (4 << 22)    // pad_interval: 32 DWORDs
          | (3 << 25);   // pad_amount:   4 DWORDs
    g1[1] = (int)((tensor_d0 & 0xffffu) << 16);            // tensor_dim0 lo
    g1[2] = (int)((tensor_d0 >> 16) | ((tensor_d1 & 0xffffu) << 16));
    g1[3] = (int)((tensor_d1 >> 16) | (64u << 16));        // tile_dim0 = 64
    g1[4] = 64;                                            // tile_dim1=64, tile_dim2=0
    g1[5] = (int)d0_stride;                                // dim0 stride lo
    g1[6] = 0;                                             // stride hi, dim1_stride lo
    g1[7] = 0;
    i32x4 gz;
    gz[0] = 0; gz[1] = 0; gz[2] = 0; gz[3] = 0;
#if __clang_major__ >= 23
    i32x8 gz8;
#pragma unroll
    for (int i = 0; i < 8; ++i) gz8[i] = 0;
    __builtin_amdgcn_tensor_load_to_lds(g0, g1, gz, gz, gz8, 0);
#else
    __builtin_amdgcn_tensor_load_to_lds(g0, g1, gz, gz, 0);
#endif
}
#endif  // USE_TDM

__device__ __forceinline__ v8f vzero8() {
    v8f z;
#pragma unroll
    for (int i = 0; i < 8; ++i) z[i] = 0.0f;
    return z;
}

// ============================================================
// f32 -> f16 conversion (vectorized, n multiple of 4)
// ============================================================
__global__ __launch_bounds__(256) void f32_to_f16_kernel(
    const float* __restrict__ src, h16* __restrict__ dst, int n4) {
    int i = blockIdx.x * 256 + threadIdx.x;
    if (i < n4) {
        f32x4 v = *(const f32x4*)(src + (size_t)i * 4);
        h16* d = dst + (size_t)i * 4;
        d[0] = (h16)v[0]; d[1] = (h16)v[1]; d[2] = (h16)v[2]; d[3] = (h16)v[3];
    }
}

// ============================================================
// GEMM: C[m,n] = sum_k A_h[m,k] * B_h[n,k] + bias[n]
// block = 128x128, 8 waves (2x4), wave tile 64x32 (4x2 WMMA tiles)
// Double-buffered LDS slabs; one barrier per k-step.
// ============================================================
template <typename OutT>
__global__ __launch_bounds__(256) void gemm_bias_kernel(
    const h16* __restrict__ A, const h16* __restrict__ Bw,
    const float* __restrict__ bias, OutT* __restrict__ Cout,
    int M, int N, int K) {
    __shared__ __align__(16) h16 As[2][128 * 48];
    __shared__ __align__(16) h16 Bs[2][128 * 48];

    const int tid  = threadIdx.x;
    const int lane = tid & 31;
    const int wid  = tid >> 5;
    const int sub  = lane & 15;
    const int sel  = lane >> 4;
    const int wm   = wid >> 2;   // 0..1
    const int wn   = wid & 3;    // 0..3
    const int mBase = blockIdx.y * 128;
    const int nBase = blockIdx.x * 128;

    v8f acc[4][2];
#pragma unroll
    for (int i = 0; i < 4; ++i)
#pragma unroll
        for (int j = 0; j < 2; ++j) acc[i][j] = vzero8();

    // prologue: stage slab 0
#pragma unroll
    for (int i = 0; i < 2; ++i) {
        int idx = tid * 2 + i;           // 0..511
        int row = idx >> 2, seg = idx & 3;
        *(f32x4*)(&As[0][row * 48 + seg * 8]) =
            *(const f32x4*)(A + (size_t)(mBase + row) * K + seg * 8);
        *(f32x4*)(&Bs[0][row * 48 + seg * 8]) =
            *(const f32x4*)(Bw + (size_t)(nBase + row) * K + seg * 8);
    }

    int cur = 0;
    for (int k0 = 0; k0 < K; k0 += 32) {
        __syncthreads();   // slab 'cur' visible; all prior reads of 'cur^1' done

        const bool hasNext = (k0 + 32 < K);
        f32x4 ra[2], rb[2];
        if (hasNext) {      // issue next slab's global loads early
#pragma unroll
            for (int i = 0; i < 2; ++i) {
                int idx = tid * 2 + i;
                int row = idx >> 2, seg = idx & 3;
                ra[i] = *(const f32x4*)(A + (size_t)(mBase + row) * K + k0 + 32 + seg * 8);
                rb[i] = *(const f32x4*)(Bw + (size_t)(nBase + row) * K + k0 + 32 + seg * 8);
            }
        }

        FragH af[4], bf[2];
#pragma unroll
        for (int i = 0; i < 4; ++i) {  // A-layout: row per lane, K halves 0..15/16..31
            int r = (wm * 64 + i * 16 + sub) * 48;
            af[i].f4[0] = *(const f32x4*)(&As[cur][r + sel * 16]);
            af[i].f4[1] = *(const f32x4*)(&As[cur][r + sel * 16 + 8]);
        }
#pragma unroll
        for (int j = 0; j < 2; ++j) {  // B-layout: col per lane, K halves 0..15/16..31
            int r = (wn * 32 + j * 16 + sub) * 48;
            bf[j].f4[0] = *(const f32x4*)(&Bs[cur][r + sel * 16]);
            bf[j].f4[1] = *(const f32x4*)(&Bs[cur][r + sel * 16 + 8]);
        }
#pragma unroll
        for (int i = 0; i < 4; ++i)
#pragma unroll
            for (int j = 0; j < 2; ++j)
                acc[i][j] = __builtin_amdgcn_wmma_f32_16x16x32_f16(
                    false, af[i].v, false, bf[j].v, (short)0, acc[i][j], false, false);

        if (hasNext) {      // commit next slab to the alternate buffer
#pragma unroll
            for (int i = 0; i < 2; ++i) {
                int idx = tid * 2 + i;
                int row = idx >> 2, seg = idx & 3;
                *(f32x4*)(&As[cur ^ 1][row * 48 + seg * 8]) = ra[i];
                *(f32x4*)(&Bs[cur ^ 1][row * 48 + seg * 8]) = rb[i];
            }
        }
        cur ^= 1;
    }

    // epilogue: C layout (lane<16: M=r, lane>=16: M=r+8; N = lane&15)
#pragma unroll
    for (int i = 0; i < 4; ++i)
#pragma unroll
        for (int j = 0; j < 2; ++j)
#pragma unroll
            for (int r = 0; r < 8; ++r) {
                int row = mBase + wm * 64 + i * 16 + sel * 8 + r;
                int col = nBase + wn * 32 + j * 16 + sub;
                float v = acc[i][j][r] + bias[col];
                Cout[(size_t)row * N + col] = (OutT)v;
            }
}

// ============================================================
// prep: read qkv (f16 [4096,3072]), L2-normalize q/k per (b,h,t),
// write Qn/Kn f16 [B,H,T,D]; write V transposed bf16 [B,H,D,T]
// one wave per (b,h,t) row
// ============================================================
__global__ __launch_bounds__(256) void prep_qkv_kernel(
    const h16* __restrict__ qkvh, h16* __restrict__ qh,
    h16* __restrict__ kh, bf16* __restrict__ vT) {
    const int lane = threadIdx.x & 31;
    const int wid  = threadIdx.x >> 5;
    const int id   = blockIdx.x * 8 + wid;   // (b*H + h)*T + t
    const int t    = id & (SEQ - 1);
    const int bhh  = id >> 11;               // b*H + h
    const int h    = bhh & (HEADS - 1);
    const int b    = bhh >> 4;
    const int m    = b * SEQ + t;
    const h16* row = qkvh + (size_t)m * NQKV;
    const int d0   = lane * 2;

    float q0 = (float)row[h * HDIM + d0];
    float q1 = (float)row[h * HDIM + d0 + 1];
    float k0 = (float)row[EMBED + h * HDIM + d0];
    float k1 = (float)row[EMBED + h * HDIM + d0 + 1];
    float v0 = (float)row[2 * EMBED + h * HDIM + d0];
    float v1 = (float)row[2 * EMBED + h * HDIM + d0 + 1];

    float sq = q0 * q0 + q1 * q1;
    float sk = k0 * k0 + k1 * k1;
#pragma unroll
    for (int mask = 1; mask < 32; mask <<= 1) {
        sq += __shfl_xor(sq, mask, 32);
        sk += __shfl_xor(sk, mask, 32);
    }
    float qs = 1.0f / fmaxf(sqrtf(sq), 1e-12f);
    float ks = 1.0f / fmaxf(sqrtf(sk), 1e-12f);

    size_t o = (size_t)id * HDIM + d0;
    qh[o]     = (h16)(q0 * qs);
    qh[o + 1] = (h16)(q1 * qs);
    kh[o]     = (h16)(k0 * ks);
    kh[o + 1] = (h16)(k1 * ks);
    vT[((size_t)bhh * HDIM + d0) * SEQ + t]     = (bf16)v0;
    vT[((size_t)bhh * HDIM + d0 + 1) * SEQ + t] = (bf16)v1;
}

// ============================================================
// YAT attention: block = 128 query rows for one (b,h); 8 waves,
// each owns 16 query rows. Stream key blocks of 64 (causal bound).
// K/V tiles staged by the Tensor Data Mover when available (one
// descriptor per tile, TENSORcnt-tracked, LDS row padding done by
// the DMA engine); otherwise cooperative vector loads.
// S = Qn*Kn^T (f16 WMMA), f = s^2/max(2+eps-2s,1e-6) causal-masked.
// numer AND z both via bf16 WMMA (V^T augmented with a ones row at
// d=64, so column 64 of f(S)@[V|1] accumulates z on the matrix core).
// out = numer/z written f16 into [B,T,C] layout.
// ============================================================
__global__ __launch_bounds__(256) void yat_attention_kernel(
    const h16* __restrict__ qh, const h16* __restrict__ kh,
    const bf16* __restrict__ vT, h16* __restrict__ x2h) {
    // Qs (128x64 f16, stride 72) reused as Ss (bf16 scores);
    // Ks 64x64 f16; Vs 80x64 bf16 (rows 64..79 = ones/zero pad)
    __shared__ __align__(16) unsigned char smem[128 * 72 * 2 + 64 * 72 * 2 + 80 * 72 * 2];
    h16*  Qs = (h16*)smem;
    bf16* Ss = (bf16*)smem;
    h16*  Ks = (h16*)(smem + 128 * 72 * 2);
    bf16* Vs = (bf16*)(smem + 128 * 72 * 2 + 64 * 72 * 2);

    const int tid  = threadIdx.x;
    const int lane = tid & 31;
    const int wid  = tid >> 5;
    const int sub  = lane & 15;
    const int sel  = lane >> 4;
    const int bh    = blockIdx.y;           // b*H + h
    const int qbase = blockIdx.x * 128;
    const h16*  Qg = qh + (size_t)bh * SEQ * HDIM;
    const h16*  Kg = kh + (size_t)bh * SEQ * HDIM;
    const bf16* Vg = vT + (size_t)bh * HDIM * SEQ;

    // stage Q tile 128x64 (1024 chunks of 16B)
#pragma unroll
    for (int i = 0; i < 4; ++i) {
        int idx = tid + i * 256;
        int row = idx >> 3, seg = idx & 7;
        *(f32x4*)(&Qs[row * 72 + seg * 8]) =
            *(const f32x4*)(Qg + (size_t)(qbase + row) * HDIM + seg * 8);
    }
    // one-time init of augmented V' rows: d=64 -> ones, d=65..79 -> zeros
#pragma unroll
    for (int i = tid; i < 16 * 64; i += 256) {
        int rr = i >> 6, cc = i & 63;
        Vs[(64 + rr) * 72 + cc] = (rr == 0) ? (bf16)1.0f : (bf16)0.0f;
    }
    __syncthreads();

    FragH qa[2];   // A fragments for D-steps 0..31, 32..63 (kept in registers)
#pragma unroll
    for (int ks = 0; ks < 2; ++ks) {
        int base = (wid * 16 + sub) * 72 + ks * 32;
        qa[ks].f4[0] = *(const f32x4*)(&Qs[base + sel * 16]);
        qa[ks].f4[1] = *(const f32x4*)(&Qs[base + sel * 16 + 8]);
    }

    auto loadK = [&](FragH* dst, int nt) {
        int kr = (nt * 16 + sub) * 72;
        dst[0].f4[0] = *(const f32x4*)(&Ks[kr + sel * 16]);
        dst[0].f4[1] = *(const f32x4*)(&Ks[kr + sel * 16 + 8]);
        dst[1].f4[0] = *(const f32x4*)(&Ks[kr + 32 + sel * 16]);
        dst[1].f4[1] = *(const f32x4*)(&Ks[kr + 32 + sel * 16 + 8]);
    };
    auto loadV = [&](FragBF* dst, int nt) {
        int vr = (nt * 16 + sub) * 72;
        dst[0].f4[0] = *(const f32x4*)(&Vs[vr + sel * 16]);
        dst[0].f4[1] = *(const f32x4*)(&Vs[vr + sel * 16 + 8]);
        dst[1].f4[0] = *(const f32x4*)(&Vs[vr + 32 + sel * 16]);
        dst[1].f4[1] = *(const f32x4*)(&Vs[vr + 32 + sel * 16 + 8]);
    };

    v8f nacc[5];   // nacc[0..3] = numer over D; nacc[4] col 0 = z
#pragma unroll
    for (int nt = 0; nt < 5; ++nt) nacc[nt] = vzero8();

    for (int kb = 0; kb < qbase + 128; kb += 64) {
        __syncthreads();   // protect Ks/Vs reuse across iterations
#if USE_TDM
        if (wid == 0) {
            // K tile: rows t=kb..kb+63, 64 contiguous halves each
            tdm_load_tile_2d(Kg + (size_t)kb * HDIM,
                             (unsigned)(size_t)(void*)Ks,
                             HDIM, SEQ, HDIM);
            // V^T tile: rows d=0..63, 64 contiguous bf16 at col kb
            tdm_load_tile_2d(Vg + kb,
                             (unsigned)(size_t)(void*)Vs,
                             SEQ, HDIM, SEQ);
            __builtin_amdgcn_s_wait_tensorcnt(0);
        }
        __syncthreads();
#else
        if (kb + 64 < qbase + 128) {   // near-cache prefetch of next key block
            int row = tid >> 3, seg = tid & 7;
            __builtin_prefetch(Kg + (size_t)(kb + 64 + row) * HDIM + seg * 8, 0, 3);
            __builtin_prefetch(Vg + (size_t)row * SEQ + kb + 64 + seg * 8, 0, 3);
        }
#pragma unroll
        for (int i = 0; i < 2; ++i) {      // stage K (64x64 f16) and V^T (64x64 bf16)
            int idx = tid + i * 256;
            int row = idx >> 3, seg = idx & 7;
            *(f32x4*)(&Ks[row * 72 + seg * 8]) =
                *(const f32x4*)(Kg + (size_t)(kb + row) * HDIM + seg * 8);
            *(f32x4*)(&Vs[row * 72 + seg * 8]) =
                *(const f32x4*)(Vg + (size_t)row * SEQ + kb + seg * 8);
        }
        __syncthreads();
#endif

        // S = Qn * Kn^T : 4 N-tiles x 2 D-steps of f16 WMMA, pipelined frags
        v8f s[4];
        FragH kf[2][2];
        loadK(kf[0], 0);
#pragma unroll
        for (int nt = 0; nt < 4; ++nt) {
            if (nt < 3) loadK(kf[(nt + 1) & 1], nt + 1);
            v8f c = vzero8();
            c = __builtin_amdgcn_wmma_f32_16x16x32_f16(
                false, qa[0].v, false, kf[nt & 1][0].v, (short)0, c, false, false);
            c = __builtin_amdgcn_wmma_f32_16x16x32_f16(
                false, qa[1].v, false, kf[nt & 1][1].v, (short)0, c, false, false);
            s[nt] = c;
        }

        // elementwise YAT transform + causal mask, scores -> LDS (bf16)
#pragma unroll
        for (int nt = 0; nt < 4; ++nt)
#pragma unroll
            for (int r = 0; r < 8; ++r) {
                int grow = qbase + wid * 16 + sel * 8 + r;
                int gcol = kb + nt * 16 + sub;
                float x = s[nt][r];
                float f = 0.0f;
                if (gcol <= grow) {
                    float den = fmaxf(CCONST - 2.0f * x, 1e-6f);
                    f = x * x / den;
                }
                Ss[(wid * 16 + sel * 8 + r) * 72 + nt * 16 + sub] = (bf16)f;
            }

        // [numer | z] += f(S) @ [V | 1] : same-wave DS ordering makes Ss
        // visible without a barrier. 5 N-tiles x 2 key-steps, pipelined.
        FragBF sa[2];
#pragma unroll
        for (int ks = 0; ks < 2; ++ks) {
            int base = (wid * 16 + sub) * 72 + ks * 32;
            sa[ks].f4[0] = *(const f32x4*)(&Ss[base + sel * 16]);
            sa[ks].f4[1] = *(const f32x4*)(&Ss[base + sel * 16 + 8]);
        }
        FragBF vf[2][2];
        loadV(vf[0], 0);
#pragma unroll
        for (int nt = 0; nt < 5; ++nt) {
            if (nt < 4) loadV(vf[(nt + 1) & 1], nt + 1);
            nacc[nt] = __builtin_amdgcn_wmma_f32_16x16x32_bf16(
                false, sa[0].v, false, vf[nt & 1][0].v, (short)0, nacc[nt], false, false);
            nacc[nt] = __builtin_amdgcn_wmma_f32_16x16x32_bf16(
                false, sa[1].v, false, vf[nt & 1][1].v, (short)0, nacc[nt], false, false);
        }
    }

    // broadcast z (column 64 of augmented GEMM -> lane sub==0 of each half)
    float zrow[8];
#pragma unroll
    for (int r = 0; r < 8; ++r)
        zrow[r] = fmaxf(__shfl(nacc[4][r], sel * 16, 32), 1e-6f);

    // epilogue: out = numer / z, scatter f16 into [B,T,C]
    const int b = bh >> 4, h = bh & (HEADS - 1);
#pragma unroll
    for (int nt = 0; nt < 4; ++nt)
#pragma unroll
        for (int r = 0; r < 8; ++r) {
            int grow = qbase + wid * 16 + sel * 8 + r;
            int d = nt * 16 + sub;
            x2h[(size_t)(b * SEQ + grow) * EMBED + h * HDIM + d] =
                (h16)(nacc[nt][r] / zrow[r]);
        }
}

// ============================================================
extern "C" void kernel_launch(void* const* d_in, const int* in_sizes, int n_in,
                              void* d_out, int out_size, void* d_ws, size_t ws_size,
                              hipStream_t stream) {
    const float* x     = (const float*)d_in[0];
    const float* w_qkv = (const float*)d_in[1];
    const float* b_qkv = (const float*)d_in[2];
    const float* w_out = (const float*)d_in[3];
    const float* b_out = (const float*)d_in[4];
    float* out = (float*)d_out;

    char* ws = (char*)d_ws;
    h16* xh   = (h16*)ws;  ws += (size_t)MROWS * EMBED * 2;
    h16* wqh  = (h16*)ws;  ws += (size_t)NQKV * EMBED * 2;
    h16* woh  = (h16*)ws;  ws += (size_t)EMBED * EMBED * 2;
    h16* qkvh = (h16*)ws;  ws += (size_t)MROWS * NQKV * 2;
    h16* qh   = (h16*)ws;  ws += (size_t)MROWS * EMBED * 2;   // [B,H,T,D]
    h16* kh   = (h16*)ws;  ws += (size_t)MROWS * EMBED * 2;   // [B,H,T,D]
    bf16* vT  = (bf16*)ws; ws += (size_t)MROWS * EMBED * 2;   // [B,H,D,T]
    h16* x2h  = (h16*)ws;  ws += (size_t)MROWS * EMBED * 2;   // [B,T,C]

    f32_to_f16_kernel<<<MROWS * EMBED / 4 / 256, 256, 0, stream>>>(x, xh, MROWS * EMBED / 4);
    f32_to_f16_kernel<<<NQKV * EMBED / 4 / 256, 256, 0, stream>>>(w_qkv, wqh, NQKV * EMBED / 4);
    f32_to_f16_kernel<<<EMBED * EMBED / 4 / 256, 256, 0, stream>>>(w_out, woh, EMBED * EMBED / 4);

    gemm_bias_kernel<h16><<<dim3(NQKV / 128, MROWS / 128), 256, 0, stream>>>(
        xh, wqh, b_qkv, qkvh, MROWS, NQKV, EMBED);

    prep_qkv_kernel<<<BATCH * HEADS * SEQ / 8, 256, 0, stream>>>(qkvh, qh, kh, vT);

    yat_attention_kernel<<<dim3(SEQ / 128, BATCH * HEADS), 256, 0, stream>>>(
        qh, kh, vT, x2h);

    gemm_bias_kernel<float><<<dim3(EMBED / 128, MROWS / 128), 256, 0, stream>>>(
        x2h, woh, b_out, out, MROWS, EMBED, EMBED);
}
